// SimplexAttention_47459388621499
// MI455X (gfx1250) — matrix operand order repaired and checked by reading, not compiled, not run on hardware
//
#include <hip/hip_runtime.h>
#include <hip/hip_bf16.h>
#include <math.h>

// ---------------------------------------------------------------------------
// SimplexAttention for MI455X (gfx1250): f16 WMMA GEMMs + fused flash attention
// GEMMs: 128x64 WG tiles, LDS double-buffered via global_load_async_to_lds_b128
// ---------------------------------------------------------------------------

#define DIM    768
#define HEADS  12
#define HEAD_D 64
#define BATCH  2
#define N1     4096     // keys per batch
#define N2     2048     // queries per batch
#define M_OUT  (BATCH * N2)   // 4096 rows of x_out
#define M_IN   (BATCH * N1)   // 8192 rows of x_in
#define DIM2   (2 * DIM)      // 1536

typedef __attribute__((ext_vector_type(16))) _Float16 v16h;
typedef __attribute__((ext_vector_type(8)))  _Float16 h8;
typedef __attribute__((ext_vector_type(8)))  float    v8f;
typedef __attribute__((ext_vector_type(4)))  int      v4i;

// CDNA5 async global->LDS path (ASYNCcnt), with portable fallback
#if defined(__has_builtin)
#if __has_builtin(__builtin_amdgcn_global_load_async_to_lds_b128)
#define SX_HAS_ASYNC 1
#endif
#endif
#ifndef SX_HAS_ASYNC
#define SX_HAS_ASYNC 0
#endif

#if SX_HAS_ASYNC
#define SX_AS1 __attribute__((address_space(1)))
#define SX_AS3 __attribute__((address_space(3)))
#endif

// -------------------------------- f32 -> f16 -------------------------------
// n must be a multiple of 8 (all buffers here are 768-based sizes).
__global__ void sx_cvt_f16(const float* __restrict__ src,
                           _Float16* __restrict__ dst, size_t n) {
  size_t i = (size_t)blockIdx.x * blockDim.x + threadIdx.x;
  const size_t stride = (size_t)gridDim.x * blockDim.x;
  const size_t n8 = n >> 3;
  const float4* s4 = (const float4*)src;
  for (; i < n8; i += stride) {
    float4 a = s4[2 * i];
    float4 b = s4[2 * i + 1];
    h8 o;
    o[0] = (_Float16)a.x; o[1] = (_Float16)a.y;
    o[2] = (_Float16)a.z; o[3] = (_Float16)a.w;
    o[4] = (_Float16)b.x; o[5] = (_Float16)b.y;
    o[6] = (_Float16)b.z; o[7] = (_Float16)b.w;
    *(h8*)(dst + 8 * i) = o;
  }
}

__device__ __forceinline__ float sx_gelu(float v) {
  return 0.5f * v * (1.0f + erff(v * 0.70710678118654752f));
}

// ------------------------------- WMMA GEMM ---------------------------------
// C[M,N] = act(A[M,K] @ W[N,K]^T + bias).  M%128==0, N%64==0, K%64==0.
// Block = 128 threads (4 waves). WG tile 128x64; wave w owns rows 32w..32w+31
// (two 16-row WMMA tiles, so each B fragment is reused twice).
// K staged in 64-wide chunks through double-buffered LDS tiles (padded rows).
#define SX_BK  64
#define SX_LDP 72   // padded LDS row stride in halfs (36 dwords: conflict-free)

__global__ __launch_bounds__(128)
void sx_gemm_wmma(const _Float16* __restrict__ A, const _Float16* __restrict__ W,
                  const float* __restrict__ bias,
                  _Float16* __restrict__ outH, float* __restrict__ outF,
                  int M, int N, int K, int act) {
  __shared__ _Float16 As[2][128][SX_LDP];
  __shared__ _Float16 Ws[2][64][SX_LDP];

  const int tid  = threadIdx.x;
  const int wave = tid >> 5;
  const int lane = tid & 31;
  const int half = lane >> 4;
  const int l16  = lane & 15;
  const int m0 = blockIdx.x * 128;
  const int n0 = blockIdx.y * 64;

  // cooperative fill: A row per thread (8 x 16B); W: half-row per thread (4 x 16B)
  const _Float16* agl = A + (size_t)(m0 + tid) * K;
  const _Float16* wgl = W + (size_t)(n0 + (tid >> 1)) * K + (tid & 1) * 32;

  const int S = K / SX_BK;
  v8f acc[2][4] = {};

  auto load_stage = [&](int s, int buf) {
    const _Float16* ag = agl + s * SX_BK;
    const _Float16* wg = wgl + s * SX_BK;
#if SX_HAS_ASYNC
#pragma unroll
    for (int i = 0; i < 8; ++i)
      __builtin_amdgcn_global_load_async_to_lds_b128(
          (SX_AS1 v4i*)(ag + i * 8),
          (SX_AS3 v4i*)(&As[buf][tid][i * 8]), 0, 0);
#pragma unroll
    for (int i = 0; i < 4; ++i)
      __builtin_amdgcn_global_load_async_to_lds_b128(
          (SX_AS1 v4i*)(wg + i * 8),
          (SX_AS3 v4i*)(&Ws[buf][tid >> 1][(tid & 1) * 32 + i * 8]), 0, 0);
#else
#pragma unroll
    for (int i = 0; i < 8; ++i)
      *(h8*)(&As[buf][tid][i * 8]) = *(const h8*)(ag + i * 8);
#pragma unroll
    for (int i = 0; i < 4; ++i)
      *(h8*)(&Ws[buf][tid >> 1][(tid & 1) * 32 + i * 8]) = *(const h8*)(wg + i * 8);
#endif
  };

  auto frag_a = [&](int buf, int row, int kk) -> v16h {
    const _Float16* ar = &As[buf][row][kk];
    h8 lo = *(const h8*)(ar + half * 16);
    h8 hi = *(const h8*)(ar + half * 16 + 8);
    v16h a;
#pragma unroll
    for (int e = 0; e < 8; ++e) { a[e] = lo[e]; a[8 + e] = hi[e]; }
    return a;
  };

  auto compute = [&](int buf) {
#pragma unroll
    for (int kk = 0; kk < SX_BK; kk += 32) {
      v16h a0 = frag_a(buf, wave * 32 + l16, kk);
      v16h a1 = frag_a(buf, wave * 32 + 16 + l16, kk);
#pragma unroll
      for (int j = 0; j < 4; ++j) {
        const _Float16* wr = &Ws[buf][16 * j + l16][kk + half * 16];
        v16h b;
        h8 lo = *(const h8*)(wr);
        h8 hi = *(const h8*)(wr + 8);
#pragma unroll
        for (int e = 0; e < 8; ++e) { b[e] = lo[e]; b[8 + e] = hi[e]; }
        acc[0][j] = __builtin_amdgcn_wmma_f32_16x16x32_f16(
            false, a0, false, b, (short)0, acc[0][j], false, false);
        acc[1][j] = __builtin_amdgcn_wmma_f32_16x16x32_f16(
            false, a1, false, b, (short)0, acc[1][j], false, false);
      }
    }
  };

  load_stage(0, 0);
  int buf = 0;
  for (int s = 0; s < S - 1; ++s) {   // steady state: next stage always in flight
    load_stage(s + 1, buf ^ 1);
#if SX_HAS_ASYNC
    asm volatile("s_wait_asynccnt 0xc" ::: "memory");
#endif
    __syncthreads();
    compute(buf);
    __syncthreads();
    buf ^= 1;
  }
#if SX_HAS_ASYNC
  asm volatile("s_wait_asynccnt 0x0" ::: "memory");
#endif
  __syncthreads();
  compute(buf);

#pragma unroll
  for (int rt = 0; rt < 2; ++rt) {
#pragma unroll
    for (int j = 0; j < 4; ++j) {
      const int col = n0 + 16 * j + l16;
      const float bv = bias ? bias[col] : 0.0f;
#pragma unroll
      for (int r = 0; r < 8; ++r) {
        const int row = m0 + wave * 32 + rt * 16 + r + half * 8;
        float v = acc[rt][j][r] + bv;
        if (act == 1) v = sx_gelu(v);
        if (outF) outF[(size_t)row * N + col] = v;
        else      outH[(size_t)row * N + col] = (_Float16)v;
      }
    }
  }
}

// --------------------------- Fused flash attention -------------------------
// grid.x = BATCH*HEADS, grid.y = N2/64.  4 waves, each owns a 16-query tile.
// Streams 32-key chunks of K/V through LDS; online softmax; ctx in f32 regs.
__global__ __launch_bounds__(128)
void sx_attn_wmma(const _Float16* __restrict__ Q, const _Float16* __restrict__ Kh,
                  const _Float16* __restrict__ Vh, _Float16* __restrict__ ctx) {
  __shared__ _Float16 Kt[32][72];      // K chunk [key][d], padded rows
  __shared__ _Float16 Vt[64][40];      // V chunk transposed [d][key], padded
  __shared__ _Float16 Pt[4][16][40];   // per-wave probs tile [q][key], padded

  const int bh  = blockIdx.x;
  const int bi  = bh / HEADS;
  const int h   = bh % HEADS;
  const int q0  = blockIdx.y * 64;
  const int wave = threadIdx.x >> 5;
  const int lane = threadIdx.x & 31;
  const int half = lane >> 4;
  const int l16  = lane & 15;

  // Q fragments (d=0..31 and d=32..63), pre-scaled by 1/sqrt(HEAD_D)
  const _Float16* qrow =
      Q + (size_t)(bi * N2 + q0 + wave * 16 + l16) * DIM + h * HEAD_D;
  v16h qa[2];
#pragma unroll
  for (int t = 0; t < 2; ++t) {
    h8 lo = *(const h8*)(qrow + 32 * t + half * 16);
    h8 hi = *(const h8*)(qrow + 32 * t + half * 16 + 8);
#pragma unroll
    for (int e = 0; e < 8; ++e) {
      qa[t][e]     = lo[e] * (_Float16)0.125f;
      qa[t][8 + e] = hi[e] * (_Float16)0.125f;
    }
  }

  v8f cacc[4] = {};
  float rmax[8], rsum[8];
#pragma unroll
  for (int r = 0; r < 8; ++r) { rmax[r] = -1.0e30f; rsum[r] = 0.0f; }

  const int trow = threadIdx.x >> 2;        // 0..31  (key row in chunk)
  const int tcol = (threadIdx.x & 3) * 16;  // 0,16,32,48 (d offset)

  for (int k0 = 0; k0 < N1; k0 += 32) {
    // cooperative stage of K (async, row-major) and V (transposed) into LDS
    {
      const size_t grow = (size_t)(bi * N1 + k0 + trow) * DIM + h * HEAD_D + tcol;
#if SX_HAS_ASYNC
      __builtin_amdgcn_global_load_async_to_lds_b128(
          (SX_AS1 v4i*)(Kh + grow), (SX_AS3 v4i*)(&Kt[trow][tcol]), 0, 0);
      __builtin_amdgcn_global_load_async_to_lds_b128(
          (SX_AS1 v4i*)(Kh + grow + 8), (SX_AS3 v4i*)(&Kt[trow][tcol + 8]), 0, 0);
#else
      h8 kv0 = *(const h8*)(Kh + grow);
      h8 kv1 = *(const h8*)(Kh + grow + 8);
      *(h8*)(&Kt[trow][tcol])     = kv0;
      *(h8*)(&Kt[trow][tcol + 8]) = kv1;
#endif
      h8 vv0 = *(const h8*)(Vh + grow);
      h8 vv1 = *(const h8*)(Vh + grow + 8);
#pragma unroll
      for (int e = 0; e < 8; ++e) {
        Vt[tcol + e][trow]     = vv0[e];
        Vt[tcol + 8 + e][trow] = vv1[e];
      }
    }
#if SX_HAS_ASYNC
    asm volatile("s_wait_asynccnt 0x0" ::: "memory");
#endif
    __syncthreads();

    // S = Q K^T for 2 x (16q x 16k) tiles, each accumulating over d=64 (2 WMMAs)
    v8f s[2];
#pragma unroll
    for (int t = 0; t < 2; ++t) {
      const _Float16* kl = &Kt[t * 16 + l16][0];
      v16h b0, b1;
      h8 x0 = *(const h8*)(kl + half * 16);
      h8 x1 = *(const h8*)(kl + half * 16 + 8);
      h8 y0 = *(const h8*)(kl + 32 + half * 16);
      h8 y1 = *(const h8*)(kl + 32 + half * 16 + 8);
#pragma unroll
      for (int e = 0; e < 8; ++e) {
        b0[e] = x0[e]; b0[8 + e] = x1[e];
        b1[e] = y0[e]; b1[8 + e] = y1[e];
      }
      v8f z = {};
      z = __builtin_amdgcn_wmma_f32_16x16x32_f16(false, qa[0], false, b0,
                                                 (short)0, z, false, false);
      z = __builtin_amdgcn_wmma_f32_16x16x32_f16(false, qa[1], false, b1,
                                                 (short)0, z, false, false);
      s[t] = z;
    }

    // online softmax: row r lives in one 16-lane half (wave32 C layout)
#pragma unroll
    for (int r = 0; r < 8; ++r) {
      float m = fmaxf(s[0][r], s[1][r]);
      m = fmaxf(m, __shfl_xor(m, 1, 32));
      m = fmaxf(m, __shfl_xor(m, 2, 32));
      m = fmaxf(m, __shfl_xor(m, 4, 32));
      m = fmaxf(m, __shfl_xor(m, 8, 32));
      const float nm    = fmaxf(rmax[r], m);
      const float scale = __expf(rmax[r] - nm);
      const float p0 = __expf(s[0][r] - nm);
      const float p1 = __expf(s[1][r] - nm);
      float ps = p0 + p1;
      ps += __shfl_xor(ps, 1, 32);
      ps += __shfl_xor(ps, 2, 32);
      ps += __shfl_xor(ps, 4, 32);
      ps += __shfl_xor(ps, 8, 32);
      rsum[r] = rsum[r] * scale + ps;
      rmax[r] = nm;
      Pt[wave][r + half * 8][l16]      = (_Float16)p0;
      Pt[wave][r + half * 8][16 + l16] = (_Float16)p1;
#pragma unroll
      for (int j = 0; j < 4; ++j) cacc[j][r] = cacc[j][r] * scale;
    }

    // ctx += P V   (A from per-wave LDS probs tile; B from transposed V tile)
    v16h pa;
    {
      const _Float16* pr = &Pt[wave][l16][0];
      h8 lo = *(const h8*)(pr + half * 16);
      h8 hi = *(const h8*)(pr + half * 16 + 8);
#pragma unroll
      for (int e = 0; e < 8; ++e) { pa[e] = lo[e]; pa[8 + e] = hi[e]; }
    }
#pragma unroll
    for (int j = 0; j < 4; ++j) {
      const _Float16* vl = &Vt[16 * j + l16][0];
      v16h vb;
      h8 lo = *(const h8*)(vl + half * 16);
      h8 hi = *(const h8*)(vl + half * 16 + 8);
#pragma unroll
      for (int e = 0; e < 8; ++e) { vb[e] = lo[e]; vb[8 + e] = hi[e]; }
      cacc[j] = __builtin_amdgcn_wmma_f32_16x16x32_f16(
          false, pa, false, vb, (short)0, cacc[j], false, false);
    }
    __syncthreads();
  }

  // finalize: ctx / rowsum -> f16 [b*N2, DIM] with head-interleaved columns
#pragma unroll
  for (int j = 0; j < 4; ++j) {
#pragma unroll
    for (int r = 0; r < 8; ++r) {
      const int row = bi * N2 + q0 + wave * 16 + r + half * 8;
      const int d   = 16 * j + l16;
      ctx[(size_t)row * DIM + h * HEAD_D + d] =
          (_Float16)(cacc[j][r] / rsum[r]);
    }
  }
}

// ------------------------- LayerNorm + FiLM epilogue -----------------------
// out = layernorm(x_out) * (gain + 1) + bias;  mod[row] = [gain(768) bias(768)]
__global__ __launch_bounds__(256)
void sx_film_ln(const float* __restrict__ xout, const float* __restrict__ mod,
                float* __restrict__ out) {
  const int row = blockIdx.x;
  const float* x = xout + (size_t)row * DIM;
  const float* g = mod + (size_t)row * DIM2;
  __shared__ float rs[8], rs2[8];

  float s = 0.0f, s2 = 0.0f;
  for (int c = threadIdx.x; c < DIM; c += 256) {
    float v = x[c]; s += v; s2 += v * v;
  }
#pragma unroll
  for (int m = 1; m < 32; m <<= 1) {
    s  += __shfl_xor(s,  m, 32);
    s2 += __shfl_xor(s2, m, 32);
  }
  if ((threadIdx.x & 31) == 0) { rs[threadIdx.x >> 5] = s; rs2[threadIdx.x >> 5] = s2; }
  __syncthreads();
  float ts = 0.0f, ts2 = 0.0f;
#pragma unroll
  for (int i = 0; i < 8; ++i) { ts += rs[i]; ts2 += rs2[i]; }
  const float mu  = ts * (1.0f / (float)DIM);
  const float var = ts2 * (1.0f / (float)DIM) - mu * mu;
  const float inv = rsqrtf(var + 1e-5f);
  for (int c = threadIdx.x; c < DIM; c += 256) {
    const float v = (x[c] - mu) * inv;
    out[(size_t)row * DIM + c] = v * (g[c] + 1.0f) + g[DIM + c];
  }
}

// -------------------------------- launcher ---------------------------------
extern "C" void kernel_launch(void* const* d_in, const int* in_sizes, int n_in,
                              void* d_out, int out_size, void* d_ws, size_t ws_size,
                              hipStream_t stream) {
  (void)in_sizes; (void)n_in; (void)out_size; (void)ws_size;

  const float* x_in  = (const float*)d_in[0];
  const float* x_out = (const float*)d_in[1];
  const float* Wq = (const float*)d_in[2];  const float* bq = (const float*)d_in[3];
  const float* Wk = (const float*)d_in[4];  const float* bk = (const float*)d_in[5];
  const float* Wv = (const float*)d_in[6];  const float* bv = (const float*)d_in[7];
  const float* Wp = (const float*)d_in[8];  const float* bp = (const float*)d_in[9];
  const float* Wm1 = (const float*)d_in[10]; const float* bm1 = (const float*)d_in[11];
  const float* Wm2 = (const float*)d_in[12]; const float* bm2 = (const float*)d_in[13];

  char* ws = (char*)d_ws;
  size_t off = 0;
  auto alloc = [&](size_t bytes) -> void* {
    void* p = ws + off;
    off += (bytes + 255) & ~(size_t)255;
    return p;
  };

  _Float16* xin_h  = (_Float16*)alloc((size_t)M_IN  * DIM * 2);
  _Float16* xout_h = (_Float16*)alloc((size_t)M_OUT * DIM * 2);
  _Float16* wq_h   = (_Float16*)alloc((size_t)DIM * DIM * 2);
  _Float16* wk_h   = (_Float16*)alloc((size_t)DIM * DIM * 2);
  _Float16* wv_h   = (_Float16*)alloc((size_t)DIM * DIM * 2);
  _Float16* wp_h   = (_Float16*)alloc((size_t)DIM * DIM * 2);
  _Float16* wm1_h  = (_Float16*)alloc((size_t)DIM2 * DIM * 2);
  _Float16* wm2_h  = (_Float16*)alloc((size_t)DIM2 * DIM2 * 2);
  _Float16* q_h    = (_Float16*)alloc((size_t)M_OUT * DIM * 2);
  _Float16* k_h    = (_Float16*)alloc((size_t)M_IN  * DIM * 2);
  _Float16* v_h    = (_Float16*)alloc((size_t)M_IN  * DIM * 2);
  _Float16* ctx_h  = (_Float16*)alloc((size_t)M_OUT * DIM * 2);
  _Float16* ctl_h  = (_Float16*)alloc((size_t)M_OUT * DIM * 2);
  _Float16* mid_h  = (_Float16*)alloc((size_t)M_OUT * DIM2 * 2);
  float*    mod_f  = (float*)   alloc((size_t)M_OUT * DIM2 * 4);

  const dim3 cblk(256);
  const dim3 cgrd(1024);
  sx_cvt_f16<<<cgrd, cblk, 0, stream>>>(x_in,  xin_h,  (size_t)M_IN  * DIM);
  sx_cvt_f16<<<cgrd, cblk, 0, stream>>>(x_out, xout_h, (size_t)M_OUT * DIM);
  sx_cvt_f16<<<cgrd, cblk, 0, stream>>>(Wq,  wq_h,  (size_t)DIM * DIM);
  sx_cvt_f16<<<cgrd, cblk, 0, stream>>>(Wk,  wk_h,  (size_t)DIM * DIM);
  sx_cvt_f16<<<cgrd, cblk, 0, stream>>>(Wv,  wv_h,  (size_t)DIM * DIM);
  sx_cvt_f16<<<cgrd, cblk, 0, stream>>>(Wp,  wp_h,  (size_t)DIM * DIM);
  sx_cvt_f16<<<cgrd, cblk, 0, stream>>>(Wm1, wm1_h, (size_t)DIM2 * DIM);
  sx_cvt_f16<<<cgrd, cblk, 0, stream>>>(Wm2, wm2_h, (size_t)DIM2 * DIM2);

  const dim3 gblk(128);
  // Q/K/V projections  (WG tile is 128 rows x 64 cols)
  sx_gemm_wmma<<<dim3(M_OUT / 128, DIM / 64), gblk, 0, stream>>>(
      xout_h, wq_h, bq, q_h, nullptr, M_OUT, DIM, DIM, 0);
  sx_gemm_wmma<<<dim3(M_IN / 128, DIM / 64), gblk, 0, stream>>>(
      xin_h, wk_h, bk, k_h, nullptr, M_IN, DIM, DIM, 0);
  sx_gemm_wmma<<<dim3(M_IN / 128, DIM / 64), gblk, 0, stream>>>(
      xin_h, wv_h, bv, v_h, nullptr, M_IN, DIM, DIM, 0);

  // fused attention
  sx_attn_wmma<<<dim3(BATCH * HEADS, N2 / 64), gblk, 0, stream>>>(
      q_h, k_h, v_h, ctx_h);

  // proj, MLP (gelu), MLP2 -> f32 modulation
  sx_gemm_wmma<<<dim3(M_OUT / 128, DIM / 64), gblk, 0, stream>>>(
      ctx_h, wp_h, bp, ctl_h, nullptr, M_OUT, DIM, DIM, 0);
  sx_gemm_wmma<<<dim3(M_OUT / 128, DIM2 / 64), gblk, 0, stream>>>(
      ctl_h, wm1_h, bm1, mid_h, nullptr, M_OUT, DIM2, DIM, 1);
  sx_gemm_wmma<<<dim3(M_OUT / 128, DIM2 / 64), gblk, 0, stream>>>(
      mid_h, wm2_h, bm2, nullptr, mod_f, M_OUT, DIM2, DIM2, 0);

  // FiLM + LayerNorm epilogue
  sx_film_ln<<<dim3(M_OUT), dim3(256), 0, stream>>>(x_out, mod_f, (float*)d_out);
}
